// GGNNRanker_13623636262995
// MI455X (gfx1250) — matrix operand order repaired and checked by reading, not compiled, not run on hardware
//
#include <hip/hip_runtime.h>
#include <hip/hip_bf16.h>

// ---------------------------------------------------------------------------
// GGNN forward for MI455X (gfx1250).
// GEMMs use V_WMMA_F32_16X16X32_F16 (f16 inputs, fp32 accumulate).
// Everything elementwise (GRU, scatter, pooling, head) is fp32.
// Workspace requirement: ~390 MB.
// ---------------------------------------------------------------------------

#define N_NODES 10000
#define N_EDGES 160000
#define IN_DIM  512
#define HID     1024
#define MP      10048      // padded node count: 157 * 64
#define NLAYERS 8

typedef __attribute__((ext_vector_type(16))) _Float16 v16h;
typedef __attribute__((ext_vector_type(8)))  _Float16 v8h;
typedef __attribute__((ext_vector_type(4)))  _Float16 v4h;
typedef __attribute__((ext_vector_type(8)))  float    v8f;

// ---------------------------------------------------------------------------
// Tiled WMMA GEMM:  C[M][N] (+)= A[M][K] * Bt[N][K]^T  (+ bias)
// A, Bt are f16 row-major; C is f32 row-major.
// Block = 256 threads = 8 waves (4 in M x 2 in N). Block tile 64x128, K step 32.
// ---------------------------------------------------------------------------
__global__ __launch_bounds__(256)
void gemm_f16_wmma(const _Float16* __restrict__ A,
                   const _Float16* __restrict__ Bt,
                   float* __restrict__ C,
                   const float* __restrict__ bias,
                   int N, int K, int accumulate)
{
    __shared__ __align__(16) _Float16 lA[64 * 40];   // 40-half row stride (80 B)
    __shared__ __align__(16) _Float16 lB[128 * 40];

    const int tid  = threadIdx.x;
    const int wave = tid >> 5;
    const int lane = tid & 31;
    const int wm   = wave & 3;    // wave M index: 16 rows each
    const int wn   = wave >> 2;   // wave N index: 64 cols each
    const int ll   = lane & 15;
    const int lhi  = lane >> 4;   // 0 or 1

    const int m0 = blockIdx.x * 64;
    const int n0 = blockIdx.y * 128;

    v8f acc[4];
#pragma unroll
    for (int t = 0; t < 4; ++t) {
        const int col = n0 + wn * 64 + t * 16 + ll;
        const float bv = bias ? bias[col] : 0.0f;
#pragma unroll
        for (int e = 0; e < 8; ++e) acc[t][e] = bv;
    }
    if (accumulate) {
#pragma unroll
        for (int t = 0; t < 4; ++t) {
            const int col  = n0 + wn * 64 + t * 16 + ll;
            const int rowb = m0 + wm * 16 + lhi * 8;
#pragma unroll
            for (int e = 0; e < 8; ++e)
                acc[t][e] += C[(size_t)(rowb + e) * N + col];
        }
    }

    for (int k0 = 0; k0 < K; k0 += 32) {
        // Stage A tile: 64 rows x 32 halves  (one b128 per thread)
        {
            const int r = tid >> 2;
            const int c = (tid & 3) * 8;
            *(v8h*)&lA[r * 40 + c] =
                *(const v8h*)(A + (size_t)(m0 + r) * K + k0 + c);
        }
        // Stage B tile: 128 rows(n) x 32 halves(k)  (two b128 per thread)
#pragma unroll
        for (int i = 0; i < 2; ++i) {
            const int n = i * 64 + (tid >> 2);
            const int c = (tid & 3) * 8;
            *(v8h*)&lB[n * 40 + c] =
                *(const v8h*)(Bt + (size_t)(n0 + n) * K + k0 + c);
        }
        __syncthreads();

        // A fragment (16x32): lane<16 -> K {0..7,16..23}; lane>=16 -> K {8..15,24..31}
        v16h af;
        {
            const int row = wm * 16 + ll;
            const int kb  = lhi * 8;
            v8h lo = *(const v8h*)&lA[row * 40 + kb];
            v8h hi = *(const v8h*)&lA[row * 40 + kb + 16];
#pragma unroll
            for (int e = 0; e < 8; ++e) { af[e] = lo[e]; af[8 + e] = hi[e]; }
        }
#pragma unroll
        for (int t = 0; t < 4; ++t) {
            // B fragment (32x16) from Bt rows: lane<16 -> K {0..7,16..23}; lane>=16 -> K {8..15,24..31}
            const int n  = wn * 64 + t * 16 + ll;
            const int kb = lhi * 8;
            v8h lo = *(const v8h*)&lB[n * 40 + kb];
            v8h hi = *(const v8h*)&lB[n * 40 + kb + 16];
            v16h bf;
#pragma unroll
            for (int e = 0; e < 8; ++e) { bf[e] = lo[e]; bf[8 + e] = hi[e]; }
            acc[t] = __builtin_amdgcn_wmma_f32_16x16x32_f16(
                false, af, false, bf, (short)0, acc[t], false, false);
        }
        __syncthreads();
    }

#pragma unroll
    for (int t = 0; t < 4; ++t) {
        const int col  = n0 + wn * 64 + t * 16 + ll;
        const int rowb = m0 + wm * 16 + lhi * 8;
#pragma unroll
        for (int e = 0; e < 8; ++e)
            C[(size_t)(rowb + e) * N + col] = acc[t][e];
    }
}

// ---------------------------------------------------------------------------
// Small helper kernels
// ---------------------------------------------------------------------------
__global__ void zero_f32(float* dst, int n4)
{
    int i = blockIdx.x * 256 + threadIdx.x;
    if (i < n4) ((float4*)dst)[i] = make_float4(0.f, 0.f, 0.f, 0.f);
}

__global__ void cvt_f32_f16(const float* __restrict__ src, _Float16* __restrict__ dst, int n4)
{
    int i = blockIdx.x * 256 + threadIdx.x;
    if (i < n4) {
        float4 v = ((const float4*)src)[i];
        v4h h;
        h[0] = (_Float16)v.x; h[1] = (_Float16)v.y;
        h[2] = (_Float16)v.z; h[3] = (_Float16)v.w;
        ((v4h*)dst)[i] = h;
    }
}

// dst[n][k] = (f16) src[k][n]   (src is [K][N] row-major)
__global__ void transpose_cvt(const float* __restrict__ src, _Float16* __restrict__ dst,
                              int K, int N)
{
    int idx = blockIdx.x * 256 + threadIdx.x;
    if (idx < K * N) {
        int k = idx / N, n = idx - k * N;
        dst[(size_t)n * K + k] = (_Float16)src[idx];
    }
}

// One block per edge: agg[dir][dst] += h[src]
__global__ __launch_bounds__(256)
void scatter_edges(const float* __restrict__ h,
                   const int* __restrict__ eidx,
                   const int* __restrict__ edir,
                   float* __restrict__ aggF,
                   float* __restrict__ aggB,
                   float* __restrict__ aggU)
{
    const int e = blockIdx.x;
    const int s = eidx[2 * e];
    const int d = eidx[2 * e + 1];
    const int dir = edir[e];
    float* dst = (dir == 0) ? aggF : (dir == 1) ? aggB : aggU;
    const float4 v = ((const float4*)(h + (size_t)s * HID))[threadIdx.x];
    float* db = dst + (size_t)d * HID + threadIdx.x * 4;
    atomicAdd(db + 0, v.x);
    atomicAdd(db + 1, v.y);
    atomicAdd(db + 2, v.z);
    atomicAdd(db + 3, v.w);
}

// messages = agg_u + b_f + b_b
__global__ void msg_init(const float* __restrict__ aggU,
                         const float* __restrict__ bf,
                         const float* __restrict__ bb,
                         float* __restrict__ msg, int total)
{
    int i = blockIdx.x * 256 + threadIdx.x;
    if (i < total) {
        int col = i & (HID - 1);
        msg[i] = aggU[i] + bf[col] + bb[col];
    }
}

// GRU cell update + relu; writes both f32 and f16 copies of h
__global__ void gru_update(const float* __restrict__ gi,
                           const float* __restrict__ gh,
                           float* __restrict__ h,
                           _Float16* __restrict__ hf, int total)
{
    int idx = blockIdx.x * 256 + threadIdx.x;
    if (idx >= total) return;
    int row = idx >> 10, col = idx & (HID - 1);
    size_t b = (size_t)row * (3 * HID) + col;
    float ir = gi[b], iz = gi[b + HID], in_ = gi[b + 2 * HID];
    float hr = gh[b], hz = gh[b + HID], hn  = gh[b + 2 * HID];
    float r = 1.0f / (1.0f + __expf(-(ir + hr)));
    float z = 1.0f / (1.0f + __expf(-(iz + hz)));
    float nn = tanhf(in_ + r * hn);
    float o = (1.0f - z) * nn + z * h[idx];
    o = o > 0.0f ? o : 0.0f;
    h[idx] = o;
    hf[idx] = (_Float16)o;
}

// column-wise max over the 10000 real nodes
__global__ void col_max(const float* __restrict__ h, float* __restrict__ out)
{
    int j = blockIdx.x * 256 + threadIdx.x;   // 4 blocks x 256 = 1024 cols
    float m = -3.4e38f;
    for (int i = 0; i < N_NODES; ++i)
        m = fmaxf(m, h[(size_t)i * HID + j]);
    out[j] = m;
}

// out = relu(g @ W_r1 + b_r1) @ W_r2 + b_r2   (single block)
__global__ __launch_bounds__(256)
void head_kernel(const float* __restrict__ g,
                 const float* __restrict__ W1, const float* __restrict__ b1,
                 const float* __restrict__ W2, const float* __restrict__ b2,
                 float* __restrict__ out)
{
    __shared__ float gs[HID];
    __shared__ float red[256];
    int tid = threadIdx.x;
    for (int j = tid; j < HID; j += 256) gs[j] = g[j];
    __syncthreads();
    float part = 0.0f;
    for (int j = tid; j < HID; j += 256) {
        float s = b1[j];
        for (int k = 0; k < HID; ++k) s += gs[k] * W1[(size_t)k * HID + j];
        s = s > 0.0f ? s : 0.0f;
        part += s * W2[j];
    }
    red[tid] = part;
    __syncthreads();
    for (int o = 128; o > 0; o >>= 1) {
        if (tid < o) red[tid] += red[tid + o];
        __syncthreads();
    }
    if (tid == 0) out[0] = red[0] + b2[0];
}

// ---------------------------------------------------------------------------
// Host orchestration
// ---------------------------------------------------------------------------
extern "C" void kernel_launch(void* const* d_in, const int* in_sizes, int n_in,
                              void* d_out, int out_size, void* d_ws, size_t ws_size,
                              hipStream_t stream)
{
    const float* node_features = (const float*)d_in[0];
    const int*   edge_index    = (const int*)d_in[1];
    const int*   edge_dirs     = (const int*)d_in[2];
    const float* W_in = (const float*)d_in[3];
    const float* b_in = (const float*)d_in[4];
    const float* W_f  = (const float*)d_in[5];
    const float* b_f  = (const float*)d_in[6];
    const float* W_b  = (const float*)d_in[7];
    const float* b_b  = (const float*)d_in[8];
    const float* W_ih = (const float*)d_in[9];
    const float* b_ih = (const float*)d_in[10];
    const float* W_hh = (const float*)d_in[11];
    const float* b_hh = (const float*)d_in[12];
    const float* W_r1 = (const float*)d_in[13];
    const float* b_r1 = (const float*)d_in[14];
    const float* W_r2 = (const float*)d_in[15];
    const float* b_r2 = (const float*)d_in[16];

    char* p = (char*)d_ws;
    auto alloc = [&](size_t bytes) -> char* {
        char* r = p;
        p += (bytes + 255) & ~(size_t)255;
        return r;
    };

    _Float16* WinT = (_Float16*)alloc((size_t)IN_DIM * HID * 2);
    _Float16* WfT  = (_Float16*)alloc((size_t)HID * HID * 2);
    _Float16* WbT  = (_Float16*)alloc((size_t)HID * HID * 2);
    _Float16* WihT = (_Float16*)alloc((size_t)HID * 3 * HID * 2);
    _Float16* WhhT = (_Float16*)alloc((size_t)HID * 3 * HID * 2);
    float*    hf32 = (float*)   alloc((size_t)MP * HID * 4);
    _Float16* hf16 = (_Float16*)alloc((size_t)MP * HID * 2);
    _Float16* Af16 = (_Float16*)alloc((size_t)MP * HID * 2);   // A-operand f16 scratch
    float*    agg  = (float*)   alloc((size_t)3 * MP * HID * 4); // agg_f/b/u; reused as gh
    float*    msg  = (float*)   alloc((size_t)MP * HID * 4);
    float*    gi   = (float*)   alloc((size_t)MP * 3 * HID * 4);
    float*    grep = (float*)   alloc((size_t)HID * 4);

    float* aggF = agg;
    float* aggB = agg + (size_t)MP * HID;
    float* aggU = agg + (size_t)2 * MP * HID;
    float* gh   = agg;                           // 3*MP*HID f32 == MP*3*HID f32

    const int totH  = MP * HID;        // elements in an [MP][HID] array
    const int blkH4 = (totH / 4 + 255) / 256;
    const int blkH  = (totH + 255) / 256;

    // --- weight conversion (transpose to [N][K] f16) ---
    transpose_cvt<<<(IN_DIM * HID + 255) / 256, 256, 0, stream>>>(W_in, WinT, IN_DIM, HID);
    transpose_cvt<<<(HID * HID + 255) / 256, 256, 0, stream>>>(W_f, WfT, HID, HID);
    transpose_cvt<<<(HID * HID + 255) / 256, 256, 0, stream>>>(W_b, WbT, HID, HID);
    transpose_cvt<<<(HID * 3 * HID + 255) / 256, 256, 0, stream>>>(W_ih, WihT, HID, 3 * HID);
    transpose_cvt<<<(HID * 3 * HID + 255) / 256, 256, 0, stream>>>(W_hh, WhhT, HID, 3 * HID);

    // --- input projection: h = nf @ W_in + b_in ---
    cvt_f32_f16<<<(N_NODES * IN_DIM / 4 + 255) / 256, 256, 0, stream>>>(
        node_features, Af16, N_NODES * IN_DIM / 4);
    {
        dim3 g(MP / 64, HID / 128);
        gemm_f16_wmma<<<g, 256, 0, stream>>>(Af16, WinT, hf32, b_in, HID, IN_DIM, 0);
    }
    cvt_f32_f16<<<blkH4, 256, 0, stream>>>(hf32, hf16, totH / 4);

    // --- message passing layers ---
    for (int layer = 0; layer < NLAYERS; ++layer) {
        zero_f32<<<(3 * totH / 4 + 255) / 256, 256, 0, stream>>>(agg, 3 * totH / 4);
        scatter_edges<<<N_EDGES, 256, 0, stream>>>(hf32, edge_index, edge_dirs,
                                                   aggF, aggB, aggU);
        msg_init<<<blkH, 256, 0, stream>>>(aggU, b_f, b_b, msg, totH);

        dim3 gN(MP / 64, HID / 128);
        dim3 g3N(MP / 64, 3 * HID / 128);

        // messages += agg_f @ W_f
        cvt_f32_f16<<<blkH4, 256, 0, stream>>>(aggF, Af16, totH / 4);
        gemm_f16_wmma<<<gN, 256, 0, stream>>>(Af16, WfT, msg, nullptr, HID, HID, 1);
        // messages += agg_b @ W_b
        cvt_f32_f16<<<blkH4, 256, 0, stream>>>(aggB, Af16, totH / 4);
        gemm_f16_wmma<<<gN, 256, 0, stream>>>(Af16, WbT, msg, nullptr, HID, HID, 1);

        // gi = messages @ W_ih + b_ih
        cvt_f32_f16<<<blkH4, 256, 0, stream>>>(msg, Af16, totH / 4);
        gemm_f16_wmma<<<g3N, 256, 0, stream>>>(Af16, WihT, gi, b_ih, 3 * HID, HID, 0);
        // gh = h @ W_hh + b_hh   (overwrites agg region; agg no longer needed)
        gemm_f16_wmma<<<g3N, 256, 0, stream>>>(hf16, WhhT, gh, b_hh, 3 * HID, HID, 0);

        gru_update<<<blkH, 256, 0, stream>>>(gi, gh, hf32, hf16, totH);
    }

    // --- readout ---
    col_max<<<HID / 256, 256, 0, stream>>>(hf32, grep);
    head_kernel<<<1, 256, 0, stream>>>(grep, W_r1, b_r1, W_r2, b_r2, (float*)d_out);
}
